// VAE_10471130267774
// MI455X (gfx1250) — compile-verified, matrix operand order checked by reading, not run on hardware
//
#include <hip/hip_runtime.h>
#include <cstddef>
#include <cstdint>

// ---------------- model constants ----------------
#define D_MODELc 512
#define D_FFc    2048
#define NHEADS   8
#define DHEAD    64
#define TTc      300
#define BBc      16
#define CFc      10      // CONTEXT_FRAMES
#define HALFW    15      // FPS/2
#define REr      (BBc * (TTc + 2))   // encoder rows = 4832
#define RDr      (BBc * TTc)         // decoder rows = 4800

// ---------------- WMMA types ----------------
typedef __bf16 bf16x16 __attribute__((ext_vector_type(16)));
typedef float  f32x8   __attribute__((ext_vector_type(8)));
typedef unsigned short u16x8 __attribute__((ext_vector_type(8)));
typedef unsigned short u16x4 __attribute__((ext_vector_type(4)));

union ABfrag { bf16x16 v; u16x8 h[2]; };
union Cfrag  { f32x8 v; float f[8]; };

__device__ __forceinline__ unsigned short f2bf(float f) {
  unsigned int x = __float_as_uint(f);
  x += 0x7FFFu + ((x >> 16) & 1u);   // round-to-nearest-even
  return (unsigned short)(x >> 16);
}

// ---------------- f32 -> bf16 convert (vectorized, n multiple of 4) --------
__global__ void k_cvt4(const float* __restrict__ x, unsigned short* __restrict__ y,
                       long long n4) {
  long long i = (long long)blockIdx.x * 256 + threadIdx.x;
  if (i >= n4) return;
  float4 v = ((const float4*)x)[i];
  u16x4 o = { f2bf(v.x), f2bf(v.y), f2bf(v.z), f2bf(v.w) };
  ((u16x4*)y)[i] = o;
}

// ---------------- f32 -> bf16 convert + transpose (K x N -> N x K) ---------
__global__ void __launch_bounds__(256)
k_cvt_t(const float* __restrict__ W, unsigned short* __restrict__ Wt,
        int K, int N) {
  __shared__ unsigned short tile[32][33];
  int k0 = blockIdx.y * 32, n0 = blockIdx.x * 32;
  int tx = threadIdx.x & 31, ty = threadIdx.x >> 5;   // 32 x 8
  #pragma unroll
  for (int r = ty; r < 32; r += 8) {
    int k = k0 + r, n = n0 + tx;
    tile[r][tx] = (k < K && n < N) ? f2bf(W[(size_t)k * N + n]) : (unsigned short)0;
  }
  __syncthreads();
  #pragma unroll
  for (int r = ty; r < 32; r += 8) {
    int n = n0 + r, k = k0 + tx;
    if (n < N && k < K) Wt[(size_t)n * K + k] = tile[tx][r];
  }
}

// ---------------- WMMA GEMM ----------------
// C[M,N] = act(A[M,K] * B[K,N] + bias) (+ resid); A bf16 row-major (MxK),
// Bt bf16 = B transposed (N x K). C f32 (+ optional bf16 copy).
#define GEMM_PRELU 1
#define GEMM_RESID 2

__global__ void __launch_bounds__(256)
k_gemm(const unsigned short* __restrict__ A,    // M x K bf16
       const unsigned short* __restrict__ Bt,   // N x K bf16
       const float* __restrict__ bias,          // N or null
       const float* __restrict__ resid,         // M x N or null
       const float* __restrict__ alphap,        // scalar or null
       float* __restrict__ C,                   // M x N
       unsigned short* __restrict__ Cbf,        // M x N bf16 or null
       int M, int N, int K, int flags) {
  __shared__ unsigned short sA[2][64][64];      // [buf][m][k]
  __shared__ unsigned short sB[2][64][64];      // [buf][n][k]
  const int tid  = threadIdx.x;
  const int wave = tid >> 5;
  const int lane = tid & 31;
  const int wm   = wave >> 1;        // 0..3  (M slab)
  const int wn   = wave & 1;         // 0..1  (N slab)
  const int half = lane >> 4;        // 0/1
  const int lm   = lane & 15;
  const int m0   = blockIdx.y * 64;
  const int n0   = blockIdx.x * 64;
  const int lr   = tid >> 2;         // 0..63 coop-load row
  const int lc   = (tid & 3) * 16;   // 0/16/32/48 coop-load col base

  f32x8 c0 = {}; f32x8 c1 = {};

  // fragments per CDNA5 16-bit A/B layout:
  //  lanes 0-15: VGPR0-3 -> K 0..7, VGPR4-7 -> K 16..23
  //  lanes16-31: VGPR0-3 -> K 8..15, VGPR4-7 -> K 24..31
  auto compute_tile = [&](int buf) {
    #pragma unroll
    for (int sub = 0; sub < 2; ++sub) {
      const int kb = sub * 32;
      ABfrag a, b0, b1;
      const int arow = wm * 16 + lm;
      a.h[0] = *(const u16x8*)&sA[buf][arow][kb + half * 8];
      a.h[1] = *(const u16x8*)&sA[buf][arow][kb + 16 + half * 8];
      const int bc0 = wn * 32 + lm;
      b0.h[0] = *(const u16x8*)&sB[buf][bc0][kb + half * 8];
      b0.h[1] = *(const u16x8*)&sB[buf][bc0][kb + 16 + half * 8];
      const int bc1 = bc0 + 16;
      b1.h[0] = *(const u16x8*)&sB[buf][bc1][kb + half * 8];
      b1.h[1] = *(const u16x8*)&sB[buf][bc1][kb + 16 + half * 8];
      c0 = __builtin_amdgcn_wmma_f32_16x16x32_bf16(false, a.v, false, b0.v,
                                                   (short)0, c0, false, false);
      c1 = __builtin_amdgcn_wmma_f32_16x16x32_bf16(false, a.v, false, b1.v,
                                                   (short)0, c1, false, false);
    }
  };

  const bool fullM = (m0 + 64 <= M);
  const bool fullN = (n0 + 64 <= N);

  if (fullM && fullN && ((K & 63) == 0)) {
    // ---------- fully pipelined, double-buffered path ----------
    const unsigned short* Aptr = A + (size_t)(m0 + lr) * K + lc;
    const unsigned short* Bptr = Bt + (size_t)(n0 + lr) * K + lc;
    u16x8 ra0 = ((const u16x8*)Aptr)[0];
    u16x8 ra1 = ((const u16x8*)Aptr)[1];
    u16x8 rb0 = ((const u16x8*)Bptr)[0];
    u16x8 rb1 = ((const u16x8*)Bptr)[1];
    *(u16x8*)&sA[0][lr][lc]     = ra0;
    *(u16x8*)&sA[0][lr][lc + 8] = ra1;
    *(u16x8*)&sB[0][lr][lc]     = rb0;
    *(u16x8*)&sB[0][lr][lc + 8] = rb1;
    __syncthreads();
    const int nk = K >> 6;
    for (int t = 0; t < nk; ++t) {
      const int cur = t & 1;
      const int nxt = cur ^ 1;
      const bool hasNext = (t + 1 < nk);
      if (hasNext) {
        // issue next tile's global loads; they complete under the WMMAs below
        const u16x8* ap = (const u16x8*)(Aptr + ((size_t)(t + 1) << 6));
        const u16x8* bp = (const u16x8*)(Bptr + ((size_t)(t + 1) << 6));
        ra0 = ap[0]; ra1 = ap[1]; rb0 = bp[0]; rb1 = bp[1];
        if (t + 2 < nk) {
          __builtin_prefetch(Aptr + ((size_t)(t + 2) << 6), 0, 1);
          __builtin_prefetch(Bptr + ((size_t)(t + 2) << 6), 0, 1);
        }
      }
      compute_tile(cur);
      if (hasNext) {
        *(u16x8*)&sA[nxt][lr][lc]     = ra0;
        *(u16x8*)&sA[nxt][lr][lc + 8] = ra1;
        *(u16x8*)&sB[nxt][lr][lc]     = rb0;
        *(u16x8*)&sB[nxt][lr][lc + 8] = rb1;
      }
      __syncthreads();
    }
  } else {
    // ---------- generic guarded path (edge tiles, odd K) ----------
    for (int k0 = 0; k0 < K; k0 += 64) {
      {
        int gm = m0 + lr;
        #pragma unroll 1
        for (int i = 0; i < 16; ++i) {
          int gk = k0 + lc + i;
          sA[0][lr][lc + i] = (gm < M && gk < K) ? A[(size_t)gm * K + gk]
                                                 : (unsigned short)0;
        }
        int gn = n0 + lr;
        #pragma unroll 1
        for (int i = 0; i < 16; ++i) {
          int gk = k0 + lc + i;
          sB[0][lr][lc + i] = (gn < N && gk < K) ? Bt[(size_t)gn * K + gk]
                                                 : (unsigned short)0;
        }
      }
      __syncthreads();
      compute_tile(0);
      __syncthreads();
    }
  }

  float alpha = ((flags & GEMM_PRELU) && alphap) ? *alphap : 0.f;
  #pragma unroll
  for (int s = 0; s < 2; ++s) {
    Cfrag cc; cc.v = s ? c1 : c0;
    int n = n0 + wn * 32 + s * 16 + lm;
    if (n >= N) continue;
    #pragma unroll
    for (int r = 0; r < 8; ++r) {
      int m = m0 + wm * 16 + half * 8 + r;  // C layout: VGPR r -> M = half*8+r
      if (m >= M) continue;
      float val = cc.f[r];
      if (bias) val += bias[n];
      if (flags & GEMM_PRELU) val = (val >= 0.f) ? val : alpha * val;
      if (flags & GEMM_RESID) val += resid[(size_t)m * N + n];
      C[(size_t)m * N + n] = val;
      if (Cbf) Cbf[(size_t)m * N + n] = f2bf(val);
    }
  }
}

// ---------------- LayerNorm (one wave32 per 512-wide row) ----------------
__global__ void __launch_bounds__(32)
k_layernorm(const float* __restrict__ x, const float* __restrict__ g,
            const float* __restrict__ b, unsigned short* __restrict__ ybf,
            int D) {
  int row  = blockIdx.x;
  int lane = threadIdx.x;
  const float4* xr = (const float4*)(x + (size_t)row * D);
  const int n4 = D >> 2;
  float s = 0.f, s2 = 0.f;
  for (int i = lane; i < n4; i += 32) {
    float4 v = xr[i];
    s  += v.x + v.y + v.z + v.w;
    s2 += v.x * v.x + v.y * v.y + v.z * v.z + v.w * v.w;
  }
  for (int o = 16; o > 0; o >>= 1) {
    s  += __shfl_xor(s,  o, 32);
    s2 += __shfl_xor(s2, o, 32);
  }
  float mean = s / D;
  float var  = s2 / D - mean * mean;
  float rs   = rsqrtf(var + 1e-5f);
  const float4* g4 = (const float4*)g;
  const float4* b4 = (const float4*)b;
  u16x4* y4 = (u16x4*)(ybf + (size_t)row * D);
  for (int i = lane; i < n4; i += 32) {
    float4 v = xr[i], gg = g4[i], bb = b4[i];
    u16x4 o = { f2bf((v.x - mean) * rs * gg.x + bb.x),
                f2bf((v.y - mean) * rs * gg.y + bb.y),
                f2bf((v.z - mean) * rs * gg.z + bb.z),
                f2bf((v.w - mean) * rs * gg.w + bb.w) };
    y4[i] = o;
  }
}

// ---------------- local windowed attention (+relative positions) ----------
// q,k,v: (B, T, 8*64) f32. tab31: (31,64) rel table for offsets -15..15.
// score(i,j) = q_i . (k_j + tab[j-i+15]) / sqrt(64) for |i-j|<=15
__global__ void __launch_bounds__(64)
k_attn(const float* __restrict__ q, const float* __restrict__ kk,
       const float* __restrict__ vv, const float* __restrict__ tab,
       float* __restrict__ out, int T, float scale) {
  __shared__ float sK[96][64];
  __shared__ float sV[96][64];
  __shared__ float sT[31][64];
  const int b = blockIdx.z, h = blockIdx.y;
  const int tile0 = blockIdx.x * 64;
  const int tid = threadIdx.x;
  const int HD = NHEADS * DHEAD;  // 512
  const size_t base = (size_t)b * T * HD + (size_t)h * DHEAD;
  const int kbase = tile0 - HALFW;
  const float4 zero4 = make_float4(0.f, 0.f, 0.f, 0.f);

  for (int r = tid; r < 94; r += 64) {
    int j = kbase + r;
    bool ok = (j >= 0 && j < T);
    const float4* kp = (const float4*)(kk + base + (size_t)(ok ? j : 0) * HD);
    const float4* vp = (const float4*)(vv + base + (size_t)(ok ? j : 0) * HD);
    float4* dk = (float4*)&sK[r][0];
    float4* dv = (float4*)&sV[r][0];
    #pragma unroll
    for (int d = 0; d < 16; ++d) {
      dk[d] = ok ? kp[d] : zero4;
      dv[d] = ok ? vp[d] : zero4;
    }
  }
  for (int r = tid; r < 31; r += 64) {
    const float4* tp = (const float4*)(tab + r * 64);
    float4* dt = (float4*)&sT[r][0];
    #pragma unroll
    for (int d = 0; d < 16; ++d) dt[d] = tp[d];
  }
  __syncthreads();

  const int i = tile0 + tid;
  if (i >= T) return;

  float qr[64];
  const float4* qp = (const float4*)(q + base + (size_t)i * HD);
  #pragma unroll
  for (int d = 0; d < 16; ++d) {
    float4 t = qp[d];
    qr[4 * d + 0] = t.x; qr[4 * d + 1] = t.y;
    qr[4 * d + 2] = t.z; qr[4 * d + 3] = t.w;
  }

  float sc[31];
  float mx = -1e30f;
  for (int w = 0; w < 31; ++w) {
    int j = i - HALFW + w;
    float s = -1e9f;
    if (j >= 0 && j < T) {
      s = 0.f;
      const int kr = tid + w;  // j - kbase
      #pragma unroll
      for (int d = 0; d < 64; ++d) s += qr[d] * (sK[kr][d] + sT[w][d]);
      s *= scale;
    }
    sc[w] = s;
    mx = fmaxf(mx, s);
  }
  float denom = 0.f;
  for (int w = 0; w < 31; ++w) { float e = __expf(sc[w] - mx); sc[w] = e; denom += e; }
  float inv = 1.f / denom;

  float4* op = (float4*)(out + base + (size_t)i * HD);
  for (int d4 = 0; d4 < 16; ++d4) {
    float a0 = 0.f, a1 = 0.f, a2 = 0.f, a3 = 0.f;
    #pragma unroll
    for (int w = 0; w < 31; ++w) {
      const float* vr = &sV[tid + w][d4 * 4];
      float s = sc[w];
      a0 += s * vr[0]; a1 += s * vr[1]; a2 += s * vr[2]; a3 += s * vr[3];
    }
    op[d4] = make_float4(a0 * inv, a1 * inv, a2 * inv, a3 * inv);
  }
}

// ---------------- small helper kernels ----------------
__global__ void k_build_enc_x(const float* __restrict__ me,
                              const float* __restrict__ mu,
                              const float* __restrict__ lv,
                              float* __restrict__ x) {
  int row = blockIdx.x;           // 0 .. RE-1
  int d   = threadIdx.x;          // 512 threads
  int t2  = row % (TTc + 2);
  int b   = row / (TTc + 2);
  float v;
  if (t2 == 0)      v = mu[d];
  else if (t2 == 1) v = lv[d];
  else              v = me[((size_t)b * TTc + (t2 - 2)) * D_MODELc + d];
  x[(size_t)row * D_MODELc + d] = v;
}

__global__ void k_relpos_in(unsigned short* __restrict__ y) {
  int i = threadIdx.x;
  if (i < 31) y[i] = f2bf((float)(i - HALFW));
}

__global__ void k_reparam(const float* __restrict__ ho, const float* __restrict__ eps,
                          float* __restrict__ z, float* __restrict__ mu_o,
                          float* __restrict__ lv_o) {
  long long i = (long long)blockIdx.x * 256 + threadIdx.x;
  long long tot = (long long)BBc * TTc * 256;
  if (i >= tot) return;
  int d = (int)(i & 255);
  long long bt = i >> 8;
  int b = (int)(bt / TTc);
  int t = (int)(bt % TTc);
  float mu = ho[((size_t)b * (TTc + 2) + 0) * 256 + d];
  float lv = ho[((size_t)b * (TTc + 2) + 1) * 256 + d];
  z[i] = mu + eps[i] * __expf(0.5f * lv);
  if (t == 0) { mu_o[b * 256 + d] = mu; lv_o[b * 256 + d] = lv; }
}

__global__ void k_build_dec_in(const float* __restrict__ motion,
                               const float* __restrict__ z,
                               unsigned short* __restrict__ mcat) {
  long long i = (long long)blockIdx.x * 256 + threadIdx.x;
  long long tot = (long long)BBc * TTc * 257;
  if (i >= tot) return;
  int d = (int)(i % 257);
  long long bt = i / 257;
  int t = (int)(bt % TTc);
  float mask = (t < CFc || t == TTc - 1) ? 1.f : 0.f;
  float v;
  if (d == 256) v = mask;
  else {
    size_t mi = (size_t)bt * 256 + d;
    v = motion[mi] * mask + z[mi] * (1.f - mask);
  }
  mcat[i] = f2bf(v);
}

__global__ void k_final_mix(const float* __restrict__ x,
                            const float* __restrict__ motion,
                            float* __restrict__ recon,
                            float* __restrict__ mask_out) {
  long long i = (long long)blockIdx.x * 256 + threadIdx.x;
  long long tot = (long long)BBc * TTc * 256;
  if (i >= tot) return;
  int t = (int)((i >> 8) % TTc);
  float mask = (t < CFc || t == TTc - 1) ? 1.f : 0.f;
  recon[i] = x[i] * (1.f - mask) + motion[i] * mask;
  if ((i & 255) == 0) mask_out[i >> 8] = mask;
}

// ---------------- host orchestration ----------------
extern "C" void kernel_launch(void* const* d_in, const int* in_sizes, int n_in,
                              void* d_out, int out_size, void* d_ws, size_t ws_size,
                              hipStream_t stream) {
  (void)in_sizes; (void)n_in; (void)out_size; (void)ws_size;

  // ---- walk params in jax pytree flatten order (sorted dict keys) ----
  int idx = 0;
  auto nf = [&]() { return (const float*)d_in[idx++]; };

  struct AttnP { const float *WkW,*Wkb,*WoW,*Wob,*WqW,*Wqb,*WvW,*Wvb,*lnb,*lng; };
  struct PffnP { const float *W1W,*W1b,*W2W,*W2b,*a,*lnb,*lng; };
  auto rdAttn = [&](AttnP& A) {
    A.WkW=nf(); A.Wkb=nf(); A.WoW=nf(); A.Wob=nf();
    A.WqW=nf(); A.Wqb=nf(); A.WvW=nf(); A.Wvb=nf();
    A.lnb=nf(); A.lng=nf();
  };
  auto rdPffn = [&](PffnP& P) {
    P.W1W=nf(); P.W1b=nf(); P.W2W=nf(); P.W2b=nf();
    P.a=nf(); P.lnb=nf(); P.lng=nf();
  };

  // params['dec'] : final_ln, head, layers, motion_enc, relpos, traj_enc
  const float *dec_fln_b=nf(), *dec_fln_g=nf();
  const float *dec_hd_a1=nf(), *dec_hd_l1W=nf(), *dec_hd_l1b=nf(),
              *dec_hd_l2W=nf(), *dec_hd_l2b=nf();
  AttnP dsa[6], dca[6]; PffnP dpf[6];
  for (int l = 0; l < 6; ++l) { rdAttn(dsa[l]); rdAttn(dca[l]); rdPffn(dpf[l]); }
  const float *dec_me_a1=nf(), *dec_me_a2=nf(),
              *dec_me_l1W=nf(), *dec_me_l1b=nf(),
              *dec_me_l2W=nf(), *dec_me_l2b=nf();
  const float *dec_rp_a1=nf(), *dec_rp_l1W=nf(), *dec_rp_l1b=nf(),
              *dec_rp_l2W=nf(), *dec_rp_l2b=nf();
  const float *dec_te_a1=nf(), *dec_te_a2=nf(),
              *dec_te_l1W=nf(), *dec_te_l1b=nf(),
              *dec_te_l2W=nf(), *dec_te_l2b=nf();

  // params['enc'] : final_ln, head, layers, logvar_tok, motion_enc, mu_tok, relpos
  const float *enc_fln_b=nf(), *enc_fln_g=nf();
  const float *enc_hd_a1=nf(), *enc_hd_l1W=nf(), *enc_hd_l1b=nf(),
              *enc_hd_l2W=nf(), *enc_hd_l2b=nf();
  AttnP esa[6]; PffnP epf[6];
  for (int l = 0; l < 6; ++l) { rdAttn(esa[l]); rdPffn(epf[l]); }
  const float *enc_lv_tok=nf();
  const float *enc_me_a1=nf(), *enc_me_a2=nf(),
              *enc_me_l1W=nf(), *enc_me_l1b=nf(),
              *enc_me_l2W=nf(), *enc_me_l2b=nf();
  const float *enc_mu_tok=nf();
  const float *enc_rp_a1=nf(), *enc_rp_l1W=nf(), *enc_rp_l1b=nf(),
              *enc_rp_l2W=nf(), *enc_rp_l2b=nf();

  const float* motion = nf();
  const float* traj   = nf();
  const float* eps    = nf();

  // ---- outputs ----
  float* recon_out = (float*)d_out;                         // B*T*256
  float* mask_out  = recon_out + (size_t)BBc * TTc * 256;   // B*T
  float* mu_out    = mask_out + (size_t)BBc * TTc;          // B*256
  float* lv_out    = mu_out + (size_t)BBc * 256;            // B*256

  // ---- workspace arena ----
  char* wsb = (char*)d_ws;
  size_t off = 0;
  auto alloc = [&](size_t bytes) -> void* {
    void* p = wsb + off;
    off += (bytes + 255) & ~(size_t)255;
    return p;
  };
  unsigned short* wbf      = (unsigned short*)alloc((size_t)D_MODELc * D_FFc * 2);
  unsigned short* rpin     = (unsigned short*)alloc(64 * 2);
  unsigned short* motionbf = (unsigned short*)alloc((size_t)RDr * 256 * 2);
  unsigned short* trajbf   = (unsigned short*)alloc((size_t)RDr * 4 * 2);
  unsigned short* mcat     = (unsigned short*)alloc((size_t)RDr * 257 * 2);
  float*          t1       = (float*)alloc((size_t)REr * D_MODELc * 4);
  unsigned short* t1bf     = (unsigned short*)alloc((size_t)REr * D_MODELc * 2);
  float*          t2       = (float*)alloc((size_t)REr * D_MODELc * 4);
  float*          xe       = (float*)alloc((size_t)REr * D_MODELc * 4);  // enc/dec x
  unsigned short* xbf      = (unsigned short*)alloc((size_t)REr * D_MODELc * 2);
  float*          ctx      = (float*)alloc((size_t)RDr * D_MODELc * 4);
  unsigned short* ctxbf    = (unsigned short*)alloc((size_t)RDr * D_MODELc * 2);
  float*          qb       = (float*)alloc((size_t)REr * D_MODELc * 4);
  float*          kb       = (float*)alloc((size_t)REr * D_MODELc * 4);
  float*          vb       = (float*)alloc((size_t)REr * D_MODELc * 4);
  float*          attno    = (float*)alloc((size_t)REr * D_MODELc * 4);
  unsigned short* attnobf  = (unsigned short*)alloc((size_t)REr * D_MODELc * 2);
  float*          ff       = (float*)alloc((size_t)REr * D_FFc * 4);
  unsigned short* ffbf     = (unsigned short*)alloc((size_t)REr * D_FFc * 2);
  float*          rph      = (float*)alloc((size_t)31 * D_MODELc * 4);
  unsigned short* rphbf    = (unsigned short*)alloc((size_t)31 * D_MODELc * 2);
  float*          tabe     = (float*)alloc((size_t)31 * 64 * 4);
  float*          tabd     = (float*)alloc((size_t)31 * 64 * 4);
  float*          heado    = (float*)alloc((size_t)REr * 256 * 4);
  float*          z        = (float*)alloc((size_t)RDr * 256 * 4);
  float*          xhat     = (float*)alloc((size_t)RDr * 256 * 4);

  // ---- launch helpers ----
  auto cvt = [&](const float* x, unsigned short* y, long long n) {
    long long n4 = n >> 2;   // all call sites have n % 4 == 0
    k_cvt4<<<(unsigned)((n4 + 255) / 256), 256, 0, stream>>>(x, y, n4);
  };
  auto gemm = [&](const unsigned short* A, const float* W, const float* bias,
                  const float* alphap, const float* resid,
                  float* C, unsigned short* Cbf, int M, int N, int K, int flags) {
    dim3 tg((unsigned)((N + 31) / 32), (unsigned)((K + 31) / 32));
    k_cvt_t<<<tg, 256, 0, stream>>>(W, wbf, K, N);   // -> Bt (N x K)
    dim3 grid((unsigned)((N + 63) / 64), (unsigned)((M + 63) / 64));
    k_gemm<<<grid, 256, 0, stream>>>(A, wbf, bias, resid, alphap, C, Cbf,
                                     M, N, K, flags);
  };
  auto lnrm = [&](const float* x, const float* g, const float* b,
                  unsigned short* ybf, int rows) {
    k_layernorm<<<(unsigned)rows, 32, 0, stream>>>(x, g, b, ybf, D_MODELc);
  };
  auto attn = [&](const float* q, const float* k, const float* v,
                  const float* tab, float* o, int T) {
    dim3 grid((unsigned)((T + 63) / 64), NHEADS, BBc);
    k_attn<<<grid, 64, 0, stream>>>(q, k, v, tab, o, T, 0.125f);
  };

  // ================= ENCODER =================
  cvt(motion, motionbf, (long long)RDr * 256);
  gemm(motionbf, enc_me_l1W, enc_me_l1b, enc_me_a1, nullptr,
       t1, t1bf, RDr, D_MODELc, 256, GEMM_PRELU);
  gemm(t1bf, enc_me_l2W, enc_me_l2b, enc_me_a2, nullptr,
       t2, nullptr, RDr, D_MODELc, D_MODELc, GEMM_PRELU);
  k_build_enc_x<<<(unsigned)REr, D_MODELc, 0, stream>>>(t2, enc_mu_tok, enc_lv_tok, xe);

  // relpos tables (enc + dec share the 31x1 input)
  k_relpos_in<<<1, 32, 0, stream>>>(rpin);
  gemm(rpin, enc_rp_l1W, enc_rp_l1b, enc_rp_a1, nullptr,
       rph, rphbf, 31, D_MODELc, 1, GEMM_PRELU);
  gemm(rphbf, enc_rp_l2W, enc_rp_l2b, nullptr, nullptr,
       tabe, nullptr, 31, 64, D_MODELc, 0);
  gemm(rpin, dec_rp_l1W, dec_rp_l1b, dec_rp_a1, nullptr,
       rph, rphbf, 31, D_MODELc, 1, GEMM_PRELU);
  gemm(rphbf, dec_rp_l2W, dec_rp_l2b, nullptr, nullptr,
       tabd, nullptr, 31, 64, D_MODELc, 0);

  for (int l = 0; l < 6; ++l) {
    const AttnP& A = esa[l];
    lnrm(xe, A.lng, A.lnb, xbf, REr);
    gemm(xbf, A.WqW, A.Wqb, nullptr, nullptr, qb, nullptr, REr, D_MODELc, D_MODELc, 0);
    gemm(xbf, A.WkW, A.Wkb, nullptr, nullptr, kb, nullptr, REr, D_MODELc, D_MODELc, 0);
    gemm(xbf, A.WvW, A.Wvb, nullptr, nullptr, vb, nullptr, REr, D_MODELc, D_MODELc, 0);
    attn(qb, kb, vb, tabe, attno, TTc + 2);
    cvt(attno, attnobf, (long long)REr * D_MODELc);
    gemm(attnobf, A.WoW, A.Wob, nullptr, xe, xe, nullptr,
         REr, D_MODELc, D_MODELc, GEMM_RESID);
    const PffnP& P = epf[l];
    lnrm(xe, P.lng, P.lnb, xbf, REr);
    gemm(xbf, P.W1W, P.W1b, P.a, nullptr, ff, ffbf, REr, D_FFc, D_MODELc, GEMM_PRELU);
    gemm(ffbf, P.W2W, P.W2b, nullptr, xe, xe, nullptr, REr, D_MODELc, D_FFc, GEMM_RESID);
  }
  lnrm(xe, enc_fln_g, enc_fln_b, xbf, REr);
  gemm(xbf, enc_hd_l1W, enc_hd_l1b, enc_hd_a1, nullptr,
       t1, t1bf, REr, D_MODELc, D_MODELc, GEMM_PRELU);
  gemm(t1bf, enc_hd_l2W, enc_hd_l2b, nullptr, nullptr,
       heado, nullptr, REr, 256, D_MODELc, 0);
  k_reparam<<<(unsigned)(((long long)RDr * 256 + 255) / 256), 256, 0, stream>>>(
      heado, eps, z, mu_out, lv_out);

  // ================= DECODER =================
  k_build_dec_in<<<(unsigned)(((long long)RDr * 257 + 255) / 256), 256, 0, stream>>>(
      motion, z, mcat);
  gemm(mcat, dec_me_l1W, dec_me_l1b, dec_me_a1, nullptr,
       t1, t1bf, RDr, D_MODELc, 257, GEMM_PRELU);
  gemm(t1bf, dec_me_l2W, dec_me_l2b, dec_me_a2, nullptr,
       xe, nullptr, RDr, D_MODELc, D_MODELc, GEMM_PRELU);
  cvt(traj, trajbf, (long long)RDr * 3);
  gemm(trajbf, dec_te_l1W, dec_te_l1b, dec_te_a1, nullptr,
       t1, t1bf, RDr, D_MODELc, 3, GEMM_PRELU);
  gemm(t1bf, dec_te_l2W, dec_te_l2b, dec_te_a2, nullptr,
       ctx, nullptr, RDr, D_MODELc, D_MODELc, GEMM_PRELU);

  for (int l = 0; l < 6; ++l) {
    // self attention
    const AttnP& A = dsa[l];
    lnrm(xe, A.lng, A.lnb, xbf, RDr);
    gemm(xbf, A.WqW, A.Wqb, nullptr, nullptr, qb, nullptr, RDr, D_MODELc, D_MODELc, 0);
    gemm(xbf, A.WkW, A.Wkb, nullptr, nullptr, kb, nullptr, RDr, D_MODELc, D_MODELc, 0);
    gemm(xbf, A.WvW, A.Wvb, nullptr, nullptr, vb, nullptr, RDr, D_MODELc, D_MODELc, 0);
    attn(qb, kb, vb, tabd, attno, TTc);
    cvt(attno, attnobf, (long long)RDr * D_MODELc);
    gemm(attnobf, A.WoW, A.Wob, nullptr, xe, xe, nullptr,
         RDr, D_MODELc, D_MODELc, GEMM_RESID);
    // cross attention (same ln params applied to x and ctx)
    const AttnP& Cx = dca[l];
    lnrm(xe,  Cx.lng, Cx.lnb, xbf,   RDr);
    lnrm(ctx, Cx.lng, Cx.lnb, ctxbf, RDr);
    gemm(xbf,   Cx.WqW, Cx.Wqb, nullptr, nullptr, qb, nullptr, RDr, D_MODELc, D_MODELc, 0);
    gemm(ctxbf, Cx.WkW, Cx.Wkb, nullptr, nullptr, kb, nullptr, RDr, D_MODELc, D_MODELc, 0);
    gemm(ctxbf, Cx.WvW, Cx.Wvb, nullptr, nullptr, vb, nullptr, RDr, D_MODELc, D_MODELc, 0);
    attn(qb, kb, vb, tabd, attno, TTc);
    cvt(attno, attnobf, (long long)RDr * D_MODELc);
    gemm(attnobf, Cx.WoW, Cx.Wob, nullptr, xe, xe, nullptr,
         RDr, D_MODELc, D_MODELc, GEMM_RESID);
    // pffn
    const PffnP& P = dpf[l];
    lnrm(xe, P.lng, P.lnb, xbf, RDr);
    gemm(xbf, P.W1W, P.W1b, P.a, nullptr, ff, ffbf, RDr, D_FFc, D_MODELc, GEMM_PRELU);
    gemm(ffbf, P.W2W, P.W2b, nullptr, xe, xe, nullptr, RDr, D_MODELc, D_FFc, GEMM_RESID);
  }
  lnrm(xe, dec_fln_g, dec_fln_b, xbf, RDr);
  gemm(xbf, dec_hd_l1W, dec_hd_l1b, dec_hd_a1, nullptr,
       t1, t1bf, RDr, D_MODELc, D_MODELc, GEMM_PRELU);
  gemm(t1bf, dec_hd_l2W, dec_hd_l2b, nullptr, nullptr,
       xhat, nullptr, RDr, 256, D_MODELc, 0);
  k_final_mix<<<(unsigned)(((long long)RDr * 256 + 255) / 256), 256, 0, stream>>>(
      xhat, motion, recon_out, mask_out);
}
